// MultiHeadAttention_12232066859249
// MI455X (gfx1250) — matrix operand order repaired and checked by reading, not compiled, not run on hardware
//
#include <hip/hip_runtime.h>
#include <hip/hip_bf16.h>
#include <hip/hip_fp16.h>

typedef __attribute__((ext_vector_type(16))) _Float16 v16h;
typedef __attribute__((ext_vector_type(8)))  _Float16 v8h;
typedef __attribute__((ext_vector_type(4)))  _Float16 v4h;
typedef __attribute__((ext_vector_type(8)))  float    v8f;

#define WMMA_F16(a, b, c) \
  __builtin_amdgcn_wmma_f32_16x16x32_f16(false, (a), false, (b), (short)0, (c), false, false)

static __device__ inline v16h cat8(v8h lo, v8h hi) {
  return __builtin_shufflevector(lo, hi, 0, 1, 2, 3, 4, 5, 6, 7, 8, 9, 10, 11, 12, 13, 14, 15);
}

// A fragment: 16x32 (MxK), f16. Lane L (L<16): row M=L, K = [0..7] in v0-3, [16..23] in v4-7.
// Lane 16+L: row M=L, K = [8..15], [24..31]. -> two 16-byte contiguous loads per lane.
static __device__ inline v16h load_frag_a(const _Float16* base, int ld, int row0, int k0) {
  const int lane = threadIdx.x & 31;
  const _Float16* p = base + (size_t)(row0 + (lane & 15)) * ld + k0 + ((lane & 16) ? 8 : 0);
  v8h lo = *(const v8h*)p;
  v8h hi = *(const v8h*)(p + 16);
  return cat8(lo, hi);
}

// B fragment: 32x16 (KxN), f16, with B[k][n] = Bt[n][k] (Bt row-major, contiguous along k).
// Lane L (L<16): column N=L, K = [0..7], [16..23]. Lane 16+L: column N=L, K = [8..15], [24..31].
static __device__ inline v16h load_frag_b(const _Float16* base, int ld, int n0, int k0) {
  const int lane = threadIdx.x & 31;
  const _Float16* p = base + (size_t)(n0 + (lane & 15)) * ld + k0 + ((lane & 16) ? 8 : 0);
  v8h lo = *(const v8h*)p;
  v8h hi = *(const v8h*)(p + 16);
  return cat8(lo, hi);
}

// ---------------------------------------------------------------------------
// f32 -> f16 conversion, 4 elements/thread
// ---------------------------------------------------------------------------
__global__ __launch_bounds__(256) void cvt_f32_f16(const float* __restrict__ in,
                                                   _Float16* __restrict__ out, int n) {
  int idx = (blockIdx.x * 256 + threadIdx.x) * 4;
  if (idx < n) {
    float4 v = *(const float4*)(in + idx);
    v4h o;
    o[0] = (_Float16)v.x;
    o[1] = (_Float16)v.y;
    o[2] = (_Float16)v.z;
    o[3] = (_Float16)v.w;
    *(v4h*)(out + idx) = o;
  }
}

// ---------------------------------------------------------------------------
// C = A (MxK, row-major f16) * Bt^T (Bt is NxK row-major f16), f32 accumulate.
// Block = 128 threads (4 waves, 2x2), block tile 64x128, wave tile 32x64.
// B fragments double-buffered in the j-loop (2 WMMAs per load); next k-step's
// A fragments fetched during the j-loop.
// MODE 0: write f16 row-major (ld=N)          -> Q, K projections
// MODE 1: write f16 per-head transposed V     -> Vt[((b*16+h)*128+d)*2048 + s]
// MODE 2: write f32 row-major (ld=N)          -> final output
// ---------------------------------------------------------------------------
template <int MODE>
__global__ __launch_bounds__(128, 4) void hgemm_nt(const _Float16* __restrict__ A,
                                                   const _Float16* __restrict__ Bt,
                                                   void* __restrict__ Out,
                                                   int M, int N, int K) {
  const int wave = threadIdx.x >> 5;
  const int lane = threadIdx.x & 31;
  const int m0 = blockIdx.x * 64 + (wave >> 1) * 32;
  const int n0 = blockIdx.y * 128 + (wave & 1) * 64;

  v8f zero = {};
  v8f acc[2][4];
#pragma unroll
  for (int i = 0; i < 2; ++i)
#pragma unroll
    for (int j = 0; j < 4; ++j) acc[i][j] = zero;

  v16h a0 = load_frag_a(A, K, m0, 0);
  v16h a1 = load_frag_a(A, K, m0 + 16, 0);

  for (int k0 = 0; k0 < K; k0 += 32) {
    if (k0 + 64 < K) {
      __builtin_prefetch(A + (size_t)(m0 + (lane & 15)) * K + k0 + 64, 0, 3);
      __builtin_prefetch(Bt + (size_t)(n0 + (lane & 15)) * K + k0 + 64, 0, 3);
      __builtin_prefetch(Bt + (size_t)(n0 + 32 + (lane & 15)) * K + k0 + 64, 0, 3);
    }
    v16h bbuf[2];
    bbuf[0] = load_frag_b(Bt, K, n0, k0);
    bbuf[1] = load_frag_b(Bt, K, n0 + 16, k0);

    v16h a0n = a0, a1n = a1;
    if (k0 + 32 < K) {
      a0n = load_frag_a(A, K, m0, k0 + 32);
      a1n = load_frag_a(A, K, m0 + 16, k0 + 32);
    }

#pragma unroll
    for (int j = 0; j < 4; ++j) {
      v16h bcur = bbuf[j & 1];
      if (j < 2) bbuf[j & 1] = load_frag_b(Bt, K, n0 + (j + 2) * 16, k0);
      acc[0][j] = WMMA_F16(a0, bcur, acc[0][j]);
      acc[1][j] = WMMA_F16(a1, bcur, acc[1][j]);
    }
    a0 = a0n;
    a1 = a1n;
  }

  const int colBase = lane & 15;
  const int rowAdd = (lane & 16) ? 8 : 0;
#pragma unroll
  for (int i = 0; i < 2; ++i)
#pragma unroll
    for (int j = 0; j < 4; ++j)
#pragma unroll
      for (int r = 0; r < 8; ++r) {
        int row = m0 + i * 16 + rowAdd + r;
        int col = n0 + j * 16 + colBase;
        float v = acc[i][j][r];
        if constexpr (MODE == 0) {
          ((_Float16*)Out)[(size_t)row * N + col] = (_Float16)v;
        } else if constexpr (MODE == 1) {
          int b = row >> 11, s = row & 2047, h = col >> 7, d = col & 127;
          ((_Float16*)Out)[(((size_t)(b * 16 + h)) * 128 + d) * 2048 + s] = (_Float16)v;
        } else {
          ((float*)Out)[(size_t)row * N + col] = v;
        }
      }
}

// ---------------------------------------------------------------------------
// Flash attention, block-cooperative LDS staging with double buffering.
// grid = (S/64, B*H), block = 128 (4 waves); each wave owns 16 query rows.
// Per iteration (32 keys): all threads issue global b128 loads for tile t+1
// into registers, compute QK/softmax/PV for tile t out of LDS (ds_load_b128
// fragments, short latency), then flush the registers to the other LDS
// buffer and barrier. Global latency is hidden by a full tile of compute.
// K tile rows padded 128->136 halfs, V tile rows 32->40 halfs to spread
// LDS banks (strides 68 / 20 dwords, both 16B-aligned).
// ---------------------------------------------------------------------------
#define KT_LD 136
#define VT_LD 40
#define KT_STRIDE (32 * KT_LD)
#define VT_STRIDE (128 * VT_LD)

__global__ __launch_bounds__(128, 4) void flash_attn(const _Float16* __restrict__ Q,
                                                     const _Float16* __restrict__ Km,
                                                     const _Float16* __restrict__ Vt,
                                                     _Float16* __restrict__ O) {
  constexpr int S = 2048, E = 2048, D = 128;
  const int bh = blockIdx.y, b = bh >> 4, h = bh & 15;
  const int tid = threadIdx.x;
  const int wave = tid >> 5, lane = tid & 31;
  const int q0 = blockIdx.x * 64 + wave * 16;
  // (1/sqrt(128)) * log2(e): softmax computed with exp2
  const float cscale = 0.12751739751000502f;

  __shared__ __align__(16) _Float16 kt[2 * KT_STRIDE];
  __shared__ __align__(16) _Float16 vtile_s[2 * VT_STRIDE];
  __shared__ __align__(16) _Float16 pbuf[4 * 16 * 32];
  _Float16* pb = pbuf + wave * (16 * 32);

  const _Float16* Qb = Q + (size_t)b * S * E + h * D;
  const _Float16* Kb = Km + (size_t)b * S * E + h * D;
  const _Float16* Vb = Vt + (size_t)bh * D * S;

  // cooperative-copy mapping: K tile 32 rows x 128 halfs, 4 threads/row x 64B
  const int krow = tid >> 2;
  const int kcol = (tid & 3) * 32;  // halfs

  // Resident Q fragments: D=128 -> 4 K-chunks of 32
  v16h qf[4];
#pragma unroll
  for (int c = 0; c < 4; ++c) qf[c] = load_frag_a(Qb, E, q0, c * 32);

  v8f zero = {};
  v8f o[8];
#pragma unroll
  for (int n = 0; n < 8; ++n) o[n] = zero;

  float mrow[8], lrow[8];
#pragma unroll
  for (int r = 0; r < 8; ++r) { mrow[r] = -1e30f; lrow[r] = 0.0f; }

  const int colBase = lane & 15;
  const int rowAdd = (lane & 16) ? 8 : 0;

  v8h kreg[4], vreg[4];

  // prologue: stage tile 0 into LDS buffer 0
  {
    const _Float16* kp = Kb + (size_t)krow * E + kcol;
    const _Float16* vp = Vb + (size_t)tid * S;
#pragma unroll
    for (int u = 0; u < 4; ++u) kreg[u] = *(const v8h*)(kp + u * 8);
#pragma unroll
    for (int u = 0; u < 4; ++u) vreg[u] = *(const v8h*)(vp + u * 8);
    _Float16* kd = kt + krow * KT_LD + kcol;
    _Float16* vd = vtile_s + tid * VT_LD;
#pragma unroll
    for (int u = 0; u < 4; ++u) *(v8h*)(kd + u * 8) = kreg[u];
#pragma unroll
    for (int u = 0; u < 4; ++u) *(v8h*)(vd + u * 8) = vreg[u];
  }
  __syncthreads();

  for (int it = 0; it < S / 32; ++it) {
    const int t0 = it * 32;
    const _Float16* ktile = kt + (it & 1) * KT_STRIDE;
    const _Float16* vtile = vtile_s + (it & 1) * VT_STRIDE;
    const bool haveNext = (t0 + 32 < S);

    // issue global loads for tile t+1; a full tile of compute hides them
    if (haveNext) {
      const _Float16* kp = Kb + (size_t)(t0 + 32 + krow) * E + kcol;
      const _Float16* vp = Vb + (size_t)tid * S + t0 + 32;
#pragma unroll
      for (int u = 0; u < 4; ++u) kreg[u] = *(const v8h*)(kp + u * 8);
#pragma unroll
      for (int u = 0; u < 4; ++u) vreg[u] = *(const v8h*)(vp + u * 8);
    }

    // ---- QK^T from LDS ----
    v8f s0 = zero, s1 = zero;
#pragma unroll
    for (int c = 0; c < 4; ++c) {
      v16h k0f = load_frag_b(ktile, KT_LD, 0, c * 32);
      v16h k1f = load_frag_b(ktile, KT_LD, 16, c * 32);
      s0 = WMMA_F16(qf[c], k0f, s0);
      s1 = WMMA_F16(qf[c], k1f, s1);
    }

    // ---- online softmax (exp2 domain) ----
    float alpha[8];
#pragma unroll
    for (int r = 0; r < 8; ++r) {
      float v0 = s0[r] * cscale;
      float v1 = s1[r] * cscale;
      float mx = fmaxf(v0, v1);
      mx = fmaxf(mx, __shfl_xor(mx, 1));
      mx = fmaxf(mx, __shfl_xor(mx, 2));
      mx = fmaxf(mx, __shfl_xor(mx, 4));
      mx = fmaxf(mx, __shfl_xor(mx, 8));
      float mnew = fmaxf(mrow[r], mx);
      float a = exp2f(mrow[r] - mnew);
      float p0 = exp2f(v0 - mnew);
      float p1 = exp2f(v1 - mnew);
      float sm = p0 + p1;
      sm += __shfl_xor(sm, 1);
      sm += __shfl_xor(sm, 2);
      sm += __shfl_xor(sm, 4);
      sm += __shfl_xor(sm, 8);
      lrow[r] = lrow[r] * a + sm;
      mrow[r] = mnew;
      alpha[r] = a;
      s0[r] = p0;
      s1[r] = p1;
    }

    // rescale accumulators
#pragma unroll
    for (int n = 0; n < 8; ++n)
#pragma unroll
      for (int r = 0; r < 8; ++r) o[n][r] *= alpha[r];

    // stage P (16x32 f16) in per-wave LDS tile: C-layout -> A-layout.
#pragma unroll
    for (int r = 0; r < 8; ++r) {
      pb[(rowAdd + r) * 32 + colBase] = (_Float16)s0[r];
      pb[(rowAdd + r) * 32 + 16 + colBase] = (_Float16)s1[r];
    }
    v16h pf = load_frag_a(pb, 32, 0, 0);

    // ---- O += P * V from LDS ----
#pragma unroll
    for (int n = 0; n < 8; ++n) {
      v16h vf = load_frag_b(vtile, VT_LD, n * 16, 0);
      o[n] = WMMA_F16(pf, vf, o[n]);
    }

    // flush staged tile t+1 into the other LDS buffer
    if (haveNext) {
      _Float16* kd = kt + ((it + 1) & 1) * KT_STRIDE + krow * KT_LD + kcol;
      _Float16* vd = vtile_s + ((it + 1) & 1) * VT_STRIDE + tid * VT_LD;
#pragma unroll
      for (int u = 0; u < 4; ++u) *(v8h*)(kd + u * 8) = kreg[u];
#pragma unroll
      for (int u = 0; u < 4; ++u) *(v8h*)(vd + u * 8) = vreg[u];
    }
    __syncthreads();
  }

  float linv[8];
#pragma unroll
  for (int r = 0; r < 8; ++r) linv[r] = 1.0f / lrow[r];
#pragma unroll
  for (int n = 0; n < 8; ++n)
#pragma unroll
    for (int r = 0; r < 8; ++r) {
      int row = q0 + rowAdd + r;
      int col = h * D + n * 16 + colBase;
      O[((size_t)b * S + row) * E + col] = (_Float16)(o[n][r] * linv[r]);
    }
}

// ---------------------------------------------------------------------------
extern "C" void kernel_launch(void* const* d_in, const int* in_sizes, int n_in,
                              void* d_out, int out_size, void* d_ws, size_t ws_size,
                              hipStream_t stream) {
  (void)in_sizes; (void)n_in; (void)out_size; (void)ws_size;
  const float* x  = (const float*)d_in[0];
  const float* Wq = (const float*)d_in[1];
  const float* Wk = (const float*)d_in[2];
  const float* Wv = (const float*)d_in[3];
  const float* Wo = (const float*)d_in[4];

  constexpr size_t XN = (size_t)4096 * 2048;  // x / q / k / v / attn elements
  constexpr size_t WN = (size_t)2048 * 2048;  // weight elements

  _Float16* p = (_Float16*)d_ws;
  _Float16* xh  = p; p += XN;
  _Float16* wqh = p; p += WN;
  _Float16* wkh = p; p += WN;
  _Float16* wvh = p; p += WN;
  _Float16* woh = p; p += WN;
  _Float16* qh  = p; p += XN;
  _Float16* kh  = p; p += XN;
  _Float16* vth = p; p += XN;
  _Float16* ah  = p; p += XN;

  // fp32 -> fp16 staging
  cvt_f32_f16<<<(XN / 4 + 255) / 256, 256, 0, stream>>>(x, xh, (int)XN);
  cvt_f32_f16<<<(WN / 4 + 255) / 256, 256, 0, stream>>>(Wq, wqh, (int)WN);
  cvt_f32_f16<<<(WN / 4 + 255) / 256, 256, 0, stream>>>(Wk, wkh, (int)WN);
  cvt_f32_f16<<<(WN / 4 + 255) / 256, 256, 0, stream>>>(Wv, wvh, (int)WN);
  cvt_f32_f16<<<(WN / 4 + 255) / 256, 256, 0, stream>>>(Wo, woh, (int)WN);

  dim3 gg(64, 16), gb(128);
  // Q = x Wq^T, K = x Wk^T (f16, row-major), V written per-head transposed
  hgemm_nt<0><<<gg, gb, 0, stream>>>(xh, wqh, qh, 4096, 2048, 2048);
  hgemm_nt<0><<<gg, gb, 0, stream>>>(xh, wkh, kh, 4096, 2048, 2048);
  hgemm_nt<1><<<gg, gb, 0, stream>>>(xh, wvh, vth, 4096, 2048, 2048);

  // attention (flash-style, LDS double-buffered K/V tiles)
  flash_attn<<<dim3(32, 32), gb, 0, stream>>>(qh, kh, vth, ah);

  // out = attn Wo^T (f32 to d_out)
  hgemm_nt<2><<<gg, gb, 0, stream>>>(ah, woh, d_out, 4096, 2048, 2048);
}
